// RWKVTimeMix_5454608466135
// MI455X (gfx1250) — matrix operand order repaired and checked by reading, not compiled, not run on hardware
//
#include <hip/hip_runtime.h>
#include <hip/hip_bf16.h>
#include <cstdint>
#include <cstddef>

typedef __attribute__((ext_vector_type(16))) __bf16 v16bf;
typedef __attribute__((ext_vector_type(8)))  __bf16 v8bf;
typedef __attribute__((ext_vector_type(4)))  __bf16 v4bf;
typedef __attribute__((ext_vector_type(8)))  float  v8f;
typedef __attribute__((ext_vector_type(4)))  float  v4f;

#define BM 128
#define BN 128
#define BK 64
#define LDK (BK + 8)   // pad to 72 bf16 (144B, 16B-aligned rows)

// Build a 16-element bf16 fragment from two 8-element (16B) LDS loads.
__device__ __forceinline__ v16bf combine8(const __bf16* p0, const __bf16* p1) {
    v8bf a = *(const v8bf*)p0;
    v8bf b = *(const v8bf*)p1;
    v16bf r;
#pragma unroll
    for (int i = 0; i < 8; ++i) { r[i] = a[i]; r[i + 8] = b[i]; }
    return r;
}

// C[M,N] = A[M,K] * W[N,K]^T   (fp32 in/out, bf16x3 WMMA inside)
__global__ __launch_bounds__(256)
void gemm_bf16x3(const float* __restrict__ Ag, const float* __restrict__ Wg,
                 float* __restrict__ Cg, int M, int N, int K) {
    __shared__ __bf16 sAh[BM][LDK];
    __shared__ __bf16 sAl[BM][LDK];
    __shared__ __bf16 sBh[BN][LDK];
    __shared__ __bf16 sBl[BN][LDK];

    const int tid  = threadIdx.x;        // 0..255, 8 waves
    const int lane = tid & 31;
    const int wave = tid >> 5;
    const int wm   = wave >> 1;          // 0..3 : wave row (32 rows each)
    const int wn   = wave & 1;           // 0..1 : wave col (64 cols each)
    const int h    = lane >> 4;          // lane half (0/1)
    const int l16  = lane & 15;
    const int m0   = blockIdx.y * BM;
    const int n0   = blockIdx.x * BN;

    v8f acc[2][4];
#pragma unroll
    for (int mi = 0; mi < 2; ++mi)
#pragma unroll
        for (int ni = 0; ni < 4; ++ni)
#pragma unroll
            for (int r = 0; r < 8; ++r) acc[mi][ni][r] = 0.0f;

    for (int k0 = 0; k0 < K; k0 += BK) {
        // ---- stage BM x BK of A and BN x BK of W into LDS as bf16 hi/lo ----
#pragma unroll
        for (int it = 0; it < (BM * BK) / (256 * 4); ++it) {
            int e   = it * 1024 + tid * 4;
            int row = e >> 6;            // /BK
            int col = e & (BK - 1);
            v4f av = *(const v4f*)(Ag + (size_t)(m0 + row) * K + k0 + col);
            v4f bv = *(const v4f*)(Wg + (size_t)(n0 + row) * K + k0 + col);
            v4bf ah, al, bh, bl;
#pragma unroll
            for (int j = 0; j < 4; ++j) {
                float xa = av[j];
                __bf16 ha = (__bf16)xa;
                ah[j] = ha;
                al[j] = (__bf16)(xa - (float)ha);
                float xb = bv[j];
                __bf16 hb = (__bf16)xb;
                bh[j] = hb;
                bl[j] = (__bf16)(xb - (float)hb);
            }
            *(v4bf*)&sAh[row][col] = ah;
            *(v4bf*)&sAl[row][col] = al;
            *(v4bf*)&sBh[row][col] = bh;
            *(v4bf*)&sBl[row][col] = bl;
        }
        __syncthreads();

#pragma unroll
        for (int kk = 0; kk < BK; kk += 32) {
            // A fragment (ISA 16-bit A 16x32 layout): m = l16,
            //   K elements: [kk + h*8, +8) then [kk + 16 + h*8, +8)
            v16bf aH[2], aL[2], bH[4], bL[4];
#pragma unroll
            for (int mi = 0; mi < 2; ++mi) {
                int row = wm * 32 + mi * 16 + l16;
                aH[mi] = combine8(&sAh[row][kk + h * 8], &sAh[row][kk + 16 + h * 8]);
                aL[mi] = combine8(&sAl[row][kk + h * 8], &sAl[row][kk + 16 + h * 8]);
            }
            // B fragment: n = l16, K elements: [kk + h*8, +8) then [kk + 16 + h*8, +8)
#pragma unroll
            for (int ni = 0; ni < 4; ++ni) {
                int row = wn * 64 + ni * 16 + l16;
                bH[ni] = combine8(&sBh[row][kk + h * 8], &sBh[row][kk + 16 + h * 8]);
                bL[ni] = combine8(&sBl[row][kk + h * 8], &sBl[row][kk + 16 + h * 8]);
            }
#pragma unroll
            for (int mi = 0; mi < 2; ++mi)
#pragma unroll
                for (int ni = 0; ni < 4; ++ni) {
                    acc[mi][ni] = __builtin_amdgcn_wmma_f32_16x16x32_bf16(
                        false, aH[mi], false, bH[ni], (short)0, acc[mi][ni], false, false);
                    acc[mi][ni] = __builtin_amdgcn_wmma_f32_16x16x32_bf16(
                        false, aH[mi], false, bL[ni], (short)0, acc[mi][ni], false, false);
                    acc[mi][ni] = __builtin_amdgcn_wmma_f32_16x16x32_bf16(
                        false, aL[mi], false, bH[ni], (short)0, acc[mi][ni], false, false);
                }
        }
        __syncthreads();
    }

    // ---- store: C/D layout: n = l16, m = r + 8*h ----
#pragma unroll
    for (int mi = 0; mi < 2; ++mi)
#pragma unroll
        for (int ni = 0; ni < 4; ++ni) {
            int mbase = m0 + wm * 32 + mi * 16 + h * 8;
            int n     = n0 + wn * 64 + ni * 16 + l16;
#pragma unroll
            for (int r = 0; r < 8; ++r)
                Cg[(size_t)(mbase + r) * N + n] = acc[mi][ni][r];
        }
}

// Fused numerically-stable WKV recurrence + sigmoid(r) gating.
// rvk layout: [B, T, 3C] with d-order r | v | k. y: [B, T, C].
__global__ __launch_bounds__(256)
void wkv_sig(const float* __restrict__ rvk, const float* __restrict__ td,
             const float* __restrict__ tf, float* __restrict__ y,
             int B_, int T_, int C_) {
    int idx = blockIdx.x * blockDim.x + threadIdx.x;
    if (idx >= B_ * C_) return;
    int b = idx / C_;
    int c = idx - b * C_;

    float w = -expf(td[c]);
    float u = tf[c];

    const float* base = rvk + (size_t)b * T_ * 3 * C_;
    float*       yb   = y   + (size_t)b * T_ * C_;

    float p = 0.0f, q = 0.0f, o = -1e38f;
    for (int t = 0; t < T_; ++t) {
        const float* row = base + (size_t)t * 3 * C_;
        float r  = row[c];
        float vv = row[C_ + c];
        float kk = row[2 * C_ + c];

        float no = fmaxf(o, u + kk);
        float A  = expf(o - no);
        float Bc = expf(u + kk - no);
        float yv = (A * p + Bc * vv) / (A * q + Bc);

        float no2 = fmaxf(w + o, kk);
        float A2  = expf(w + o - no2);
        float B2  = expf(kk - no2);
        p = A2 * p + B2 * vv;
        q = A2 * q + B2;
        o = no2;

        float sr = 1.0f / (1.0f + expf(-r));
        yb[(size_t)t * C_ + c] = sr * yv;
    }
}

extern "C" void kernel_launch(void* const* d_in, const int* in_sizes, int n_in,
                              void* d_out, int out_size, void* d_ws, size_t ws_size,
                              hipStream_t stream) {
    const float* x     = (const float*)d_in[0];   // [B,T,C]
    const float* W_rvk = (const float*)d_in[1];   // [3C,C]
    const float* W_out = (const float*)d_in[2];   // [C,C]
    const float* td    = (const float*)d_in[3];   // [C]
    const float* tf    = (const float*)d_in[4];   // [C]
    float*       out   = (float*)d_out;           // [B,T,C]

    const int C  = in_sizes[3];                   // 2048
    const int M  = in_sizes[0] / C;               // B*T = 8192
    const int N1 = in_sizes[1] / C;               // 3C = 6144
    const int N2 = in_sizes[2] / C;               // C  = 2048
    const int K  = C;
    const int T  = 1024;                          // per reference (T_MAX)
    const int B  = M / T;

    float* rvk = (float*)d_ws;                    // [M, 3C]  (192 MB)
    float* y   = rvk + (size_t)M * N1;            // [M, C]   (64 MB)

    dim3 blk(256);

    // rvk = x @ W_rvk^T
    gemm_bf16x3<<<dim3(N1 / BN, M / BM), blk, 0, stream>>>(x, W_rvk, rvk, M, N1, K);

    // y = sigmoid(r) * wkv(k, v)
    wkv_sig<<<dim3((B * C + 255) / 256), blk, 0, stream>>>(rvk, td, tf, y, B, T, C);

    // out = y @ W_out^T
    gemm_bf16x3<<<dim3(N2 / BN, M / BM), blk, 0, stream>>>(y, W_out, out, M, N2, K);
}
